// MOERouter_Optimized_8873402433831
// MI455X (gfx1250) — matrix operand order repaired and checked by reading, not compiled, not run on hardware
//
#include <hip/hip_runtime.h>
#include <hip/hip_bf16.h>
#include <math.h>

typedef __attribute__((ext_vector_type(16))) __bf16 v16bf;
typedef __attribute__((ext_vector_type(8)))  float  v8f;

// ---- problem constants (match reference) ----
#define NB 128
#define NH 1024
#define NL 512
#define NC 64        // conv out channels
#define ND 1024
#define NE 8
#define CAPACITY 24  // int(128*1.5/8)

// ---- workspace layout (bytes). total ~3.96 MB ----
#define OFF_ACC    ((size_t)0)          // [0]=regw accumulator (zeroed each launch)
#define OFF_CONVW  ((size_t)256)                         // 64*3072 bf16
#define OFF_FC1W   (OFF_CONVW + (size_t)64*3072*2)       // 128*1024 bf16
#define OFF_POOLED (OFF_FC1W  + (size_t)128*1024*2)      // 128*1024 f32 (zeroed: atomic accum)
#define OFF_HPOOL  (OFF_POOLED + (size_t)128*1024*4)     // 128*1024 f32
#define OFF_MU     (OFF_HPOOL + (size_t)128*1024*4)      // 1024 f32
#define OFF_VAR    (OFF_MU    + (size_t)1024*4)          // 1024 f32
#define OFF_HNORM  (OFF_VAR   + (size_t)1024*4)          // 128*1024 bf16
#define OFF_H1     (OFF_HNORM + (size_t)128*1024*2)      // 128*128 f32
#define OFF_RW     (OFF_H1    + (size_t)128*128*4)       // 128*8 f32
#define OFF_USAGE  (OFF_RW    + (size_t)128*8*4)         // 8 f32
#define OFF_CNT    (OFF_USAGE + (size_t)8*4)             // 8 f32
#define OFF_CW     (OFF_CNT   + (size_t)8*4)             // 128*8 f32
#define OFF_U      (OFF_CW    + (size_t)128*8*4)         // 128*8192 bf16

__device__ __forceinline__ __bf16 f2bf(float f) {
  unsigned u = __builtin_bit_cast(unsigned, f);
  u += 0x7FFFu + ((u >> 16) & 1u);                 // RNE truncate to bf16
  unsigned short s = (unsigned short)(u >> 16);
  return __builtin_bit_cast(__bf16, s);
}
__device__ __forceinline__ float gelu_exact(float v) {
  return 0.5f * v * (1.0f + erff(v * 0.70710678118654752440f));
}

// ---------------- K0: weight precision conversion ----------------
__global__ __launch_bounds__(256) void k_prep(const float* __restrict__ convw,
                                              const float* __restrict__ fc1w,
                                              __bf16* convw_bf, __bf16* fc1w_bf) {
  int i = blockIdx.x * 256 + threadIdx.x;
  if (i < 64 * 3072)  convw_bf[i] = f2bf(convw[i]);
  if (i < 128 * 1024) fc1w_bf[i]  = f2bf(fc1w[i]);
}

// ---------------- K1: conv1d(k=3,p=1)+GELU+pool16 (WMMA) + pooled mean of x ---------
// 2 blocks per sample; each block owns 256 output columns (+2-col halo).
// Conv K-dim is tap-major: k = dl*32 + hl  ->  each wmma k-step is one fixed tap.
// x tile is stored TRANSPOSED in LDS (xT[col][ch]) so B fragments (fastest index = k = ch)
// are 16 contiguous bf16 per lane -> 2x ds_load_b128, zero bank conflicts (stride 80B).
#define CST 40    // bf16 elems per column slot: 32 channels + pad (80 B, 16-B aligned frags)
__global__ __launch_bounds__(512, 1) void k_conv(const float* __restrict__ x,
                                                 const __bf16* __restrict__ convw_bf,
                                                 const float* __restrict__ convb,
                                                 float* pooled, float* hpool) {
  __shared__ __bf16 xT[258 * CST];        // [local col 0..257][in-channel 0..31]; halo col 0/257
  __shared__ __bf16 wlds[64 * 96];        // 64 out-ch x 96 (k = dl*32 + hl ordering)
  __shared__ float  poolbuf[8 * 64];      // 8 local pool buckets x out-channel
  __shared__ float  rowsum[32];
  const int b = blockIdx.x >> 1;
  const int half = blockIdx.x & 1;
  const int colbase = half * 256;
  const int t = threadIdx.x, lane = t & 31, wave = t >> 5;
  for (int i = t; i < 8 * 64; i += 512) poolbuf[i] = 0.f;
  v8f acc[4];                             // 4 M-tiles x 1 N-tile per wave
  for (int q = 0; q < 4; ++q) for (int j = 0; j < 8; ++j) acc[q][j] = 0.f;
  const int r = t >> 4;                   // loader channel row 0..31

  for (int h0 = 0; h0 < 1024; h0 += 32) {
    if (t < 32) rowsum[t] = 0.f;
    __syncthreads();
    {
      const float* xrow = x + (((size_t)b * 1024 + (h0 + r)) * 512);
      float s = 0.f;
      for (int jj = (t & 15); jj < 258; jj += 16) {   // local cols incl. halo
        int g = colbase - 1 + jj;
        float v = (g >= 0 && g < 512) ? xrow[g] : 0.f;
        if (jj >= 1 && jj < 257) s += v;              // own 256 cols only
        xT[jj * CST + r] = f2bf(v);                   // transposed store
      }
      atomicAdd(&rowsum[r], s);
      for (int i = t; i < 64 * 96; i += 512) {        // permute weights tap-major
        int c = i / 96, kk = i - c * 96;
        int dl = kk >> 5, hl = kk & 31;
        wlds[i] = convw_bf[c * 3072 + (h0 + hl) * 3 + dl];
      }
    }
    __syncthreads();
    if (t < 32) atomicAdd(&pooled[(size_t)b * 1024 + h0 + t], rowsum[t] * (1.f / 512.f));

    for (int dl = 0; dl < 3; ++dl) {      // 3 wmma k-steps of 32 (one tap each)
      v16bf bfrag;
      {
        int lc = wave * 16 + (lane & 15); // local output column
        int base = (lc + dl) * CST + ((lane >> 4) << 4);  // B layout: k = 16*(lane>>4)+i
        #pragma unroll
        for (int i = 0; i < 16; ++i) bfrag[i] = xT[base + i];   // contiguous -> 2x b128
      }
      const int m  = lane & 15;
      const int k8 = (lane >> 4) << 4;    // A layout: k = 16*(lane>>4) + i
      #pragma unroll
      for (int mt = 0; mt < 4; ++mt) {
        v16bf afrag;
        int wbase = (mt * 16 + m) * 96 + dl * 32 + k8;
        #pragma unroll
        for (int i = 0; i < 16; ++i) afrag[i] = wlds[wbase + i];
        acc[mt] = __builtin_amdgcn_wmma_f32_16x16x32_bf16(false, afrag, false, bfrag, (short)0, acc[mt], false, false);
      }
    }
  }
  __syncthreads();
  // bias + exact GELU + pool: wave owns 16 cols = half of local bucket (wave>>1)
  for (int mt = 0; mt < 4; ++mt) {
    #pragma unroll
    for (int j = 0; j < 8; ++j) {
      int c = mt * 16 + ((lane >> 4) << 3) + j;   // C layout row
      float v = acc[mt][j] + convb[c];
      atomicAdd(&poolbuf[(wave >> 1) * 64 + c], gelu_exact(v));
    }
  }
  __syncthreads();
  if (t < 8 * 64) {
    int pl = t >> 6, c = t & 63;
    // flat feature index = c*POOL + p, p = half*8 + pl
    hpool[(size_t)b * 1024 + c * 16 + half * 8 + pl] = poolbuf[pl * 64 + c] * (1.f / 32.f);
  }
}

// ---------------- K2a: batch-norm statistics (biased var) ----------------
__global__ __launch_bounds__(256) void k_bnstats(const float* __restrict__ hpool, float* mu, float* var) {
  int f = blockIdx.x * 256 + threadIdx.x;
  if (f >= 1024) return;
  float s = 0.f, s2 = 0.f;
  for (int b = 0; b < 128; ++b) { float v = hpool[b * 1024 + f]; s += v; s2 += v * v; }
  float m = s * (1.f / 128.f);
  mu[f] = m; var[f] = s2 * (1.f / 128.f) - m * m;
}

// ---------------- K2a2: normalize -> bf16 ----------------
__global__ __launch_bounds__(256) void k_bnnorm(const float* __restrict__ hpool,
                                                const float* __restrict__ mu, const float* __restrict__ var,
                                                const float* __restrict__ gamma, const float* __restrict__ beta,
                                                __bf16* hnorm) {
  int i = blockIdx.x * 256 + threadIdx.x;
  if (i >= 128 * 1024) return;
  int f = i & 1023;
  float v = (hpool[i] - mu[f]) * rsqrtf(var[f] + 1e-5f) * gamma[f] + beta[f];
  hnorm[i] = f2bf(v);
}

// ---------------- K2b: fc1 (128x128x1024 bf16 WMMA) + GELU ----------------
__global__ __launch_bounds__(512, 1) void k_fc1(const __bf16* __restrict__ hnorm,
                                                const __bf16* __restrict__ fc1w_bf,
                                                const float* __restrict__ fc1b, float* h1) {
  const int t = threadIdx.x, lane = t & 31, wave = t >> 5;
  const int mt = wave & 7, ntb = (wave >> 3) * 4;
  v8f acc[4];
  for (int nt = 0; nt < 4; ++nt) for (int j = 0; j < 8; ++j) acc[nt][j] = 0.f;
  const int m = mt * 16 + (lane & 15);
  for (int s = 0; s < 32; ++s) {
    v16bf afrag;
    #pragma unroll
    for (int i = 0; i < 16; ++i) {
      int k = s * 32 + ((lane >> 4) << 4) + i;
      afrag[i] = hnorm[m * 1024 + k];
    }
    #pragma unroll
    for (int nt = 0; nt < 4; ++nt) {
      int n = (ntb + nt) * 16 + (lane & 15);
      v16bf bfrag;
      #pragma unroll
      for (int i = 0; i < 16; ++i) {
        int k = s * 32 + ((lane >> 4) << 4) + i;
        bfrag[i] = fc1w_bf[n * 1024 + k];    // B(k,n) = fc1_w[n,k]
      }
      acc[nt] = __builtin_amdgcn_wmma_f32_16x16x32_bf16(false, afrag, false, bfrag, (short)0, acc[nt], false, false);
    }
  }
  for (int nt = 0; nt < 4; ++nt) {
    int n = (ntb + nt) * 16 + (lane & 15);
    #pragma unroll
    for (int j = 0; j < 8; ++j) {
      int mm = mt * 16 + ((lane >> 4) << 3) + j;
      h1[mm * 128 + n] = gelu_exact(acc[nt][j] + fc1b[n]);
    }
  }
}

// ---------------- K2c: fc2+softmax, dynamic-k, capacity, combine weights ----------------
__global__ __launch_bounds__(256) void k_router(const float* __restrict__ h1,
                                                const float* __restrict__ fc2w, const float* __restrict__ fc2b,
                                                float* rw_ws, float* usage_ws, float* counts_ws, float* cw_ws) {
  __shared__ float flat_w[1024];
  __shared__ int   flat_i[1024];
  __shared__ unsigned char keepf[1024];
  __shared__ float usageL[8];
  __shared__ int   cntL[8];
  __shared__ int   maxkL;
  const int t = threadIdx.x;
  if (t < 8) { usageL[t] = 0.f; cntL[t] = 0; }
  if (t == 0) maxkL = 1;
  __syncthreads();
  float tw[8]; int ti[8];
  if (t < 128) {
    const int b = t;
    float p[8];
    float mx = -1e30f;
    for (int e = 0; e < 8; ++e) {
      float d = fc2b[e];
      for (int k = 0; k < 128; ++k) d += h1[b * 128 + k] * fc2w[e * 128 + k];
      p[e] = d; mx = fmaxf(mx, d);
    }
    float sm = 0.f;
    for (int e = 0; e < 8; ++e) { p[e] = expf(p[e] - mx); sm += p[e]; }
    for (int e = 0; e < 8; ++e) { p[e] /= sm; rw_ws[b * 8 + e] = p[e]; atomicAdd(&usageL[e], p[e]); }
    bool used[8] = {false, false, false, false, false, false, false, false};
    for (int s = 0; s < 8; ++s) {          // full descending sort, stable (earliest index wins ties)
      int be = 0; float bv = -1.f;
      for (int e = 0; e < 8; ++e) if (!used[e] && p[e] > bv) { bv = p[e]; be = e; }
      used[be] = true; tw[s] = bv; ti[s] = be;
    }
    float cum = 0.f; int kp = 8;
    for (int s = 0; s < 8; ++s) { cum += tw[s]; if (cum > 0.8f) { kp = s + 1; break; } }
    atomicMax(&maxkL, kp);
  }
  __syncthreads();
  const int maxk = maxkL;
  if (t < 128)
    for (int s = 0; s < 8; ++s) { flat_w[t * 8 + s] = tw[s]; flat_i[t * 8 + s] = (s < maxk) ? ti[s] : -1; }
  __syncthreads();
  for (int i = t; i < 1024; i += 256) {    // per-expert capacity ranking (stable desc)
    int e = flat_i[i]; unsigned char kp = 0;
    if (e >= 0) {
      float wi = flat_w[i]; int rank = 0;
      for (int j = 0; j < 1024; ++j)
        if (flat_i[j] == e) { float wj = flat_w[j]; if (wj > wi || (wj == wi && j < i)) rank++; }
      if (rank < CAPACITY) { kp = 1; atomicAdd(&cntL[e], 1); }
    }
    keepf[i] = kp;
  }
  __syncthreads();
  if (t < 128) {
    const int b = t;
    float den = 0.f;
    for (int s = 0; s < maxk; ++s) den += expf(keepf[b * 8 + s] ? tw[s] : 0.0f);  // dropped slots -> logit 0
    float cw[8] = {0, 0, 0, 0, 0, 0, 0, 0};
    for (int s = 0; s < maxk; ++s) if (keepf[b * 8 + s]) cw[ti[s]] += expf(tw[s]) / den;
    for (int e = 0; e < 8; ++e) cw_ws[b * 8 + e] = cw[e];
  }
  if (t < 8) usage_ws[t] = usageL[t] * (1.f / 128.f);
  __syncthreads();
  if (t < 8) counts_ws[t] = (float)cntL[t];
}

// ---------------- K2d: u[b,(e,h)] = cw[b,e] * pooled[b,h]  (bf16) ----------------
__global__ __launch_bounds__(256) void k_buildu(const float* __restrict__ cw, const float* __restrict__ pooled,
                                                __bf16* u) {
  int i = blockIdx.x * 256 + threadIdx.x;       // < 128*8192
  int b = i >> 13, r = i & 8191, e = r >> 10, h = r & 1023;
  u[i] = f2bf(cw[b * 8 + e] * pooled[b * 1024 + h]);
}

// ---------------- K3: expert GEMM [128x8192]x[8192x1024] (WMMA) + bias + fused sum(W^2) ----
__global__ __launch_bounds__(512, 1) void k_expert(const __bf16* __restrict__ u,
                                                   const float* __restrict__ expw,
                                                   const float* __restrict__ expb,
                                                   const float* __restrict__ cw,
                                                   float* out, float* regw) {
  const int t = threadIdx.x, lane = t & 31, wave = t >> 5;
  const int ncolbase = blockIdx.x * 64;        // 16 blocks x 64 output cols
  const int mt = wave & 7, ntb = (wave >> 3) * 2;
  v8f acc[2];
  for (int nt = 0; nt < 2; ++nt) for (int j = 0; j < 8; ++j) acc[nt][j] = 0.f;
  float ssq = 0.f;
  const int mrow = mt * 16 + (lane & 15);
  const int khA = ((lane >> 4) << 3), khB = ((lane >> 4) << 4);
  for (int s = 0; s < 256; ++s) {
    if (s < 255) {  // prefetch next k-step of the streamed exp_w (-> global_prefetch_b8)
      int dpf = ncolbase + ntb * 16 + (lane & 15);
      __builtin_prefetch(&expw[(size_t)(s * 32 + 32 + khB) * 1024 + dpf], 0, 1);
    }
    v16bf afrag;
    #pragma unroll
    for (int i = 0; i < 16; ++i) {
      int k = s * 32 + khB + i;
      afrag[i] = u[(size_t)mrow * 8192 + k];
    }
    #pragma unroll
    for (int nt = 0; nt < 2; ++nt) {
      int d = ncolbase + (ntb + nt) * 16 + (lane & 15);
      v16bf bfrag;
      #pragma unroll
      for (int i = 0; i < 16; ++i) {
        int k = s * 32 + khB + i;
        float v = expw[(size_t)k * 1024 + d];  // exp_w[(e,h), d]; each element read by all 8 M-tile waves of the block
        ssq += v * v;
        bfrag[i] = f2bf(v);
      }
      acc[nt] = __builtin_amdgcn_wmma_f32_16x16x32_bf16(false, afrag, false, bfrag, (short)0, acc[nt], false, false);
    }
  }
  atomicAdd(regw, ssq);
  for (int nt = 0; nt < 2; ++nt) {
    int d = ncolbase + (ntb + nt) * 16 + (lane & 15);
    #pragma unroll
    for (int j = 0; j < 8; ++j) {
      int b = mt * 16 + ((lane >> 4) << 3) + j;
      float bias = 0.f;
      #pragma unroll
      for (int e = 0; e < 8; ++e) bias += cw[b * 8 + e] * expb[e * 1024 + d];
      out[(size_t)b * 1024 + d] = acc[nt][j] + bias;
    }
  }
}

// ---------------- K4: finalize loss / usage / counts ----------------
__global__ __launch_bounds__(256) void k_final(const float* __restrict__ expb,
                                               const float* __restrict__ usage_ws,
                                               const float* __restrict__ counts_ws,
                                               const float* __restrict__ regw, float* out) {
  __shared__ float red[256];
  const int t = threadIdx.x;
  float s = 0.f;
  for (int i = t; i < 8 * 1024; i += 256) { float v = expb[i]; s += v * v; }
  red[t] = s; __syncthreads();
  for (int off = 128; off > 0; off >>= 1) { if (t < off) red[t] += red[t + off]; __syncthreads(); }
  if (t == 0) {
    float reg = regw[0] + red[0];
    float mu = 0.f; for (int e = 0; e < 8; ++e) mu += usage_ws[e];
    mu *= 0.125f;
    float va = 0.f; for (int e = 0; e < 8; ++e) { float d = usage_ws[e] - mu; va += d * d; }
    va *= (1.f / 7.f);                       // torch.var unbiased
    out[131072] = 0.005f * va + 1e-6f * reg;
    for (int e = 0; e < 8; ++e) { out[131073 + e] = usage_ws[e]; out[131081 + e] = counts_ws[e]; }
  }
}

extern "C" void kernel_launch(void* const* d_in, const int* in_sizes, int n_in,
                              void* d_out, int out_size, void* d_ws, size_t ws_size,
                              hipStream_t stream) {
  (void)in_sizes; (void)n_in; (void)out_size; (void)ws_size;
  const float* x     = (const float*)d_in[0];
  const float* convw = (const float*)d_in[1];
  const float* convb = (const float*)d_in[2];
  const float* gamma = (const float*)d_in[3];
  const float* beta  = (const float*)d_in[4];
  const float* fc1w  = (const float*)d_in[5];
  const float* fc1b  = (const float*)d_in[6];
  const float* fc2w  = (const float*)d_in[7];
  const float* fc2b  = (const float*)d_in[8];
  const float* expw  = (const float*)d_in[9];
  const float* expb  = (const float*)d_in[10];
  char* ws = (char*)d_ws;
  float* out = (float*)d_out;

  float*  regw     = (float*)(ws + OFF_ACC);
  __bf16* convw_bf = (__bf16*)(ws + OFF_CONVW);
  __bf16* fc1w_bf  = (__bf16*)(ws + OFF_FC1W);
  float*  pooled   = (float*)(ws + OFF_POOLED);
  float*  hpool    = (float*)(ws + OFF_HPOOL);
  float*  mu       = (float*)(ws + OFF_MU);
  float*  var      = (float*)(ws + OFF_VAR);
  __bf16* hnorm    = (__bf16*)(ws + OFF_HNORM);
  float*  h1       = (float*)(ws + OFF_H1);
  float*  rw_ws    = (float*)(ws + OFF_RW);
  float*  usage_ws = (float*)(ws + OFF_USAGE);
  float*  cnt_ws   = (float*)(ws + OFF_CNT);
  float*  cw_ws    = (float*)(ws + OFF_CW);
  __bf16* u        = (__bf16*)(ws + OFF_U);

  hipMemsetAsync(ws + OFF_ACC, 0, 256, stream);                  // zero accumulators
  hipMemsetAsync(ws + OFF_POOLED, 0, (size_t)128 * 1024 * 4, stream); // pooled is atomic-accumulated
  k_prep   <<<768, 256, 0, stream>>>(convw, fc1w, convw_bf, fc1w_bf);
  k_conv   <<<256, 512, 0, stream>>>(x, convw_bf, convb, pooled, hpool);
  k_bnstats<<<4,   256, 0, stream>>>(hpool, mu, var);
  k_bnnorm <<<512, 256, 0, stream>>>(hpool, mu, var, gamma, beta, hnorm);
  k_fc1    <<<1,   512, 0, stream>>>(hnorm, fc1w_bf, fc1b, h1);
  k_router <<<1,   256, 0, stream>>>(h1, fc2w, fc2b, rw_ws, usage_ws, cnt_ws, cw_ws);
  k_buildu <<<4096,256, 0, stream>>>(cw_ws, pooled, u);
  k_expert <<<16,  512, 0, stream>>>(u, expw, expb, cw_ws, out, regw);
  k_final  <<<1,   256, 0, stream>>>(expb, usage_ws, cnt_ws, regw, out);
}
